// TemporalLinkLayer_8821862826160
// MI455X (gfx1250) — matrix operand order repaired and checked by reading, not compiled, not run on hardware
//
#include <hip/hip_runtime.h>

// ---------------------------------------------------------------------------
// TemporalLinkLayer for MI455X (gfx1250, wave32, WMMA + async-to-LDS gather)
//   logits[b] = Wfc . [relu([fu,dt_u]@W1^T + b1), relu([fv,dt_v]@W1^T + b1)] + bfc
// Pipeline per 128-edge tile (1 workgroup = 8 waves):
//   1. lane-coalesced eid load, then each wave immediately fires 32 async
//      512B row copies (GLOBAL_LOAD_ASYNC_TO_LDS_B128, ASYNCcnt-tracked,
//      no destination VGPRs) -> the random-gather HBM traffic is in flight
//      from the first cycles of the block.
//   2. under that shadow: timestamp gather + dt, W1 -> bf16 LDS conversion.
//   3. s_wait_asynccnt 0, barrier, then 64 V_WMMA_F32_16X16X32_BF16 per wave
//      (bf16 inputs, fp32 accumulate); dt rank-1 term + bias + ReLU + final
//      dot in fp32 VALU; shfl-reduce over the 16 N-lanes; direct store.
// ---------------------------------------------------------------------------

typedef __bf16 bf16_t;
typedef __attribute__((ext_vector_type(16))) __bf16 v16bf;
typedef __attribute__((ext_vector_type(8)))  float  v8f;

#define DFEAT 128
#define TILE  128   // batch rows per workgroup

extern "C" __global__ __launch_bounds__(256)
void tgn_link_wmma_async_kernel(const float* __restrict__ src_feat,
                                const float* __restrict__ dst_feat,
                                const float* __restrict__ timestamp,
                                const int*   __restrict__ src_eids,
                                const int*   __restrict__ dst_eids,
                                const float* __restrict__ t,
                                const float* __restrict__ W1,      // [128][129]
                                const float* __restrict__ b1,      // [128]
                                const float* __restrict__ Wfc,     // [256]
                                const float* __restrict__ bfc,     // [1]
                                float*       __restrict__ out,     // [B]
                                int B)
{
  extern __shared__ unsigned char smem[];
  float*  sFeat   = (float*)smem;                          // [2][128][128] fp32 (128 KB)
  bf16_t* sW1     = (bf16_t*)(sFeat + 2 * TILE * DFEAT);   // [128][128] bf16   (32 KB)
  float*  sW1last = (float*)(sW1 + DFEAT * DFEAT);         // [128]  (W1[:,128])
  float*  sB1     = sW1last + DFEAT;                       // [128]
  float*  sWfc    = sB1 + DFEAT;                           // [256]
  float*  sDt     = sWfc + 2 * DFEAT;                      // [2][128]

  const int tid  = threadIdx.x;
  const int wave = tid >> 5;          // 0..7
  const int lane = tid & 31;
  const int half = lane >> 4;         // K-half selector in WMMA layouts
  const int l15  = lane & 15;
  const int tileBase = blockIdx.x * TILE;

  // ---- wave-exclusive gather ownership: wave w owns 32 rows ----------------
  // waves 0..3 -> side u (src), rows (w&3)*32 ..; waves 4..7 -> side v (dst).
  const int side_w = wave >> 2;
  const int idx_w  = (wave & 3) * 32 + lane;     // this lane's row id 0..127
  const int b_w    = tileBase + idx_w;
  const float* feat_w = side_w ? dst_feat : src_feat;

  int eid_l = 0;
  if (b_w < B) eid_l = side_w ? dst_eids[b_w] : src_eids[b_w];   // coalesced

  // ---- fire the whole gather immediately: 32 async 512B row copies ---------
  // lane L moves bytes [16L, 16L+16) of each row; eid broadcast via readlane.
  #pragma unroll 4
  for (int r = 0; r < 32; ++r) {
    int eid = __shfl(eid_l, r);
    int idx = (wave & 3) * 32 + r;
    unsigned long long ga =
        (unsigned long long)(const void*)(feat_w + (long)eid * DFEAT) +
        (unsigned long long)(lane * 16);
    unsigned la =
        (unsigned)(unsigned long long)(const void*)(sFeat + (side_w * TILE + idx) * DFEAT) +
        (unsigned)(lane * 16);
    asm volatile("global_load_async_to_lds_b128 %0, %1, off"
                 :: "v"(la), "v"(ga) : "memory");
  }

  // ---- overlapped with the async gather: dt + W1 conversion ----------------
  {
    float dt = 0.f;
    if (b_w < B) dt = t[b_w] - timestamp[eid_l];   // random 4B gather
    sDt[side_w * TILE + idx_w] = dt;
  }
  for (int idx = tid; idx < DFEAT * DFEAT; idx += 256) {
    int n = idx >> 7, k = idx & 127;
    sW1[idx] = (bf16_t)W1[n * (DFEAT + 1) + k];    // L2-resident, 66 KB
  }
  if (tid < DFEAT) { sW1last[tid] = W1[tid * (DFEAT + 1) + DFEAT]; sB1[tid] = b1[tid]; }
  if (tid < 2 * DFEAT) sWfc[tid] = Wfc[tid];

  asm volatile("s_wait_asynccnt 0x0" ::: "memory");
  __syncthreads();

  // ---- WMMA compute: wave w owns batch rows [w*16, w*16+16) for u AND v ----
  float lacc[8];
  #pragma unroll
  for (int g = 0; g < 8; ++g) lacc[g] = 0.f;

  #pragma unroll 1
  for (int side = 0; side < 2; ++side) {
    // Build A fragments (16x32 bf16) from fp32 LDS rows, converting in regs.
    // A layout: lane half h holds K {ks*32+h*16 .. +15} (contiguous, matches B).
    const float* arow = sFeat + (size_t)(side * TILE + wave * 16 + l15) * DFEAT;
    v16bf a[4];
    #pragma unroll
    for (int ks = 0; ks < 4; ++ks) {
      const float4* p0 = (const float4*)(arow + ks * 32 + half * 16);
      const float4* p1 = (const float4*)(arow + ks * 32 + half * 16 + 8);
      float4 x0 = p0[0], x1 = p0[1];
      float4 x2 = p1[0], x3 = p1[1];
      v16bf av;
      av[0]  = (bf16_t)x0.x; av[1]  = (bf16_t)x0.y; av[2]  = (bf16_t)x0.z; av[3]  = (bf16_t)x0.w;
      av[4]  = (bf16_t)x1.x; av[5]  = (bf16_t)x1.y; av[6]  = (bf16_t)x1.z; av[7]  = (bf16_t)x1.w;
      av[8]  = (bf16_t)x2.x; av[9]  = (bf16_t)x2.y; av[10] = (bf16_t)x2.z; av[11] = (bf16_t)x2.w;
      av[12] = (bf16_t)x3.x; av[13] = (bf16_t)x3.y; av[14] = (bf16_t)x3.z; av[15] = (bf16_t)x3.w;
      a[ks] = av;
    }
    float dtv[8];
    #pragma unroll
    for (int g = 0; g < 8; ++g)
      dtv[g] = sDt[side * TILE + wave * 16 + half * 8 + g];   // C-layout M rows

    #pragma unroll 2                   // cap unroll: keep live B-fragments low
    for (int nt = 0; nt < 8; ++nt) {
      int n = nt * 16 + l15;           // this lane's output column
      const bf16_t* brow = sW1 + n * DFEAT + half * 16;  // W1 row n == W1^T col n
      v8f acc = {};
      #pragma unroll
      for (int ks = 0; ks < 4; ++ks) {
        v16bf bfrag = *(const v16bf*)(brow + ks * 32);   // 16 contiguous K values
        acc = __builtin_amdgcn_wmma_f32_16x16x32_bf16(
            false, a[ks], false, bfrag, (short)0, acc, false, false);
      }
      float wfcv = sWfc[side * DFEAT + n];
      float w1l  = sW1last[n];
      float bb   = sB1[n];
      #pragma unroll
      for (int g = 0; g < 8; ++g) {    // C: VGPR g -> M = half*8 + g
        float val = acc[g] + dtv[g] * w1l + bb;
        val = fmaxf(val, 0.f);         // ReLU
        lacc[g] = fmaf(val, wfcv, lacc[g]);
      }
    }
  }

  // ---- reduce over the 16 N-lanes of each half, then store -----------------
  #pragma unroll
  for (int g = 0; g < 8; ++g) {
    float v = lacc[g];
    v += __shfl_xor(v, 1);
    v += __shfl_xor(v, 2);
    v += __shfl_xor(v, 4);
    v += __shfl_xor(v, 8);
    lacc[g] = v;
  }
  if (l15 == 0) {
    float bv = bfc[0];
    int base = tileBase + wave * 16 + half * 8;
    #pragma unroll
    for (int g = 0; g < 8; ++g) {
      int o = base + g;
      if (o < B) out[o] = lacc[g] + bv;
    }
  }
}

extern "C" void kernel_launch(void* const* d_in, const int* in_sizes, int n_in,
                              void* d_out, int out_size, void* d_ws, size_t ws_size,
                              hipStream_t stream) {
  const float* src_feat  = (const float*)d_in[0];
  const float* dst_feat  = (const float*)d_in[1];
  const float* timestamp = (const float*)d_in[2];
  const int*   src_eids  = (const int*)  d_in[3];
  const int*   dst_eids  = (const int*)  d_in[4];
  const float* t         = (const float*)d_in[5];
  const float* W1        = (const float*)d_in[6];
  const float* b1        = (const float*)d_in[7];
  const float* Wfc       = (const float*)d_in[8];
  const float* bfc       = (const float*)d_in[9];
  float*       out       = (float*)d_out;

  const int B = in_sizes[3];
  const int tiles = (B + TILE - 1) / TILE;

  size_t smem = (size_t)(2 * TILE * DFEAT) * sizeof(float)        // fp32 feature staging
              + (size_t)(DFEAT * DFEAT) * sizeof(bf16_t)          // W1 bf16
              + (size_t)(DFEAT + DFEAT + 2 * DFEAT + 2 * TILE) * sizeof(float);

  tgn_link_wmma_async_kernel<<<tiles, 256, smem, stream>>>(
      src_feat, dst_feat, timestamp, src_eids, dst_eids, t,
      W1, b1, Wfc, bfc, out, B);
}
